// GNNNodeClassifier_32693291057615
// MI455X (gfx1250) — compile-verified, matrix operand order checked by reading, not run on hardware
//
#include <hip/hip_runtime.h>
#include <hip/hip_fp16.h>
#include <math.h>

// ---------------------------------------------------------------------------
// MI455X (gfx1250) GNN node classifier, round 2.
//   - BN + f32->f16 folded into N-sized elementwise passes (not E-sized).
//   - Weights pre-converted to f16, transposed [DOUT][DIN], zero-padded.
//   - GEMM: A fragments loaded straight from global f16 rows (gather-friendly,
//     L2-resident), B fragments from a 32KB LDS weight chunk. No barriers in
//     the K loop. v_wmma_f32_16x16x32_f16 throughout.
//   - Edge prepare-FFN fuses gather -> GEMM -> gelu -> *ew -> atomic scatter.
// ---------------------------------------------------------------------------

typedef __attribute__((ext_vector_type(16))) _Float16 v16h;
typedef __attribute__((ext_vector_type(8)))  _Float16 v8h;
typedef __attribute__((ext_vector_type(8)))  float    v8f;
typedef __attribute__((ext_vector_type(4)))  float    v4f;

#define D_FEAT 128

__device__ __forceinline__ float gelu_exact(float x) {
    return 0.5f * x * (1.0f + erff(x * 0.707106781186547524f));
}

// EPI: 0 = store gelu(acc+bias)
//      1 = atomicAdd(red[seg[row]][col], gelu(acc+bias) * ew_norm[row])
//      2 = store acc+bias (col < NCOL guard; logits head)
template<int DIN, int DOUT, bool GATHER, int EPI>
__global__ __launch_bounds__(256) void gemm_wmma(
    const _Float16* __restrict__ Ah,   // f16 activations [*, DIN]
    const int*      __restrict__ gidx,
    const _Float16* __restrict__ Wt,   // f16 weights, transposed [DOUT][DIN]
    const float*    __restrict__ bias,
    float*          __restrict__ OUT,
    const int*      __restrict__ seg_idx, const float* __restrict__ ew,
    const float*    __restrict__ ew_sum, int M, int NCOL)
{
    constexpr int ROWS = 128;        // rows per workgroup (8 waves x 16)
    constexpr int NT   = DOUT / 16;  // 16-col tiles per wave
    constexpr int KB   = 128;        // K chunk of weights held in LDS (32KB max)

    __shared__ alignas(16) _Float16 Ws[DOUT * KB];
    __shared__ int   rowg[ROWS];
    __shared__ int   segs[(EPI == 1) ? ROWS : 1];
    __shared__ float ewn [(EPI == 1) ? ROWS : 1];

    const int tid       = threadIdx.x;
    const int block_row = blockIdx.x * ROWS;

    float inv_sum = 0.0f;
    if (EPI == 1) inv_sum = 1.0f / ew_sum[0];

    for (int r = tid; r < ROWS; r += 256) {
        int gr  = block_row + r;
        int src = 0;
        if (gr < M) src = GATHER ? gidx[gr] : gr;
        rowg[r] = src;
        if (GATHER && gr < M)   // warm WGP$/L2 for the scattered gather
            __builtin_prefetch(Ah + (size_t)src * DIN, 0, 3);
        if (EPI == 1) {
            segs[r] = (gr < M) ? seg_idx[gr] : 0;
            ewn[r]  = (gr < M) ? ew[gr] * inv_sum : 0.0f;
        }
    }

    const int wave = tid >> 5;
    const int lane = tid & 31;
    const int lm   = lane & 15;
    const int lh   = lane >> 4;

    v8f acc[NT];
    #pragma unroll
    for (int t = 0; t < NT; ++t)
        #pragma unroll
        for (int i = 0; i < 8; ++i) acc[t][i] = 0.0f;

    for (int kb = 0; kb < DIN; kb += KB) {
        // Stage weight chunk: Ws[n][0..KB) = Wt[n][kb..kb+KB). Pure f16 copy.
        __syncthreads();
        for (int i = tid * 8; i < DOUT * KB; i += 256 * 8) {
            int n = i / KB, k = i % KB;
            *(v8h*)&Ws[i] = *(const v8h*)&Wt[(size_t)n * DIN + kb + k];
        }
        __syncthreads();

        const _Float16* arow =
            Ah + (size_t)rowg[wave * 16 + lm] * DIN + kb + lh * 8;

        #pragma unroll
        for (int ks = 0; ks < KB; ks += 32) {
            // A fragment (16x32 f16) direct from global: lane<16 holds
            // K 0-7 & 16-23, lane>=16 holds K 8-15 & 24-31 (ISA layout).
            v8h alo = *(const v8h*)(arow + ks);
            v8h ahi = *(const v8h*)(arow + ks + 16);
            v16h a;
            #pragma unroll
            for (int i = 0; i < 8; ++i) { a[i] = alo[i]; a[i + 8] = ahi[i]; }

            #pragma unroll
            for (int t = 0; t < NT; ++t) {
                const _Float16* brow = &Ws[(t * 16 + lm) * KB + ks + lh * 8];
                v8h blo = *(const v8h*)brow;
                v8h bhi = *(const v8h*)(brow + 16);
                v16h b;
                #pragma unroll
                for (int i = 0; i < 8; ++i) { b[i] = blo[i]; b[i + 8] = bhi[i]; }
                acc[t] = __builtin_amdgcn_wmma_f32_16x16x32_f16(
                    false, a, false, b, (short)0, acc[t], false, false);
            }
        }
    }

    // Epilogue. C layout: lane%16 = N col, VGPR v = row v (+8 for lane>=16).
    #pragma unroll
    for (int t = 0; t < NT; ++t) {
        int col  = t * 16 + lm;
        float bi = (col < NCOL) ? bias[col] : 0.0f;
        #pragma unroll
        for (int v = 0; v < 8; ++v) {
            int mloc = wave * 16 + v + lh * 8;
            int gr   = block_row + mloc;
            if (gr >= M) continue;
            float val = acc[t][v] + bi;
            if (EPI == 0) {
                OUT[(size_t)gr * NCOL + col] = gelu_exact(val);
            } else if (EPI == 1) {
                float msg = gelu_exact(val) * ewn[mloc];
                atomicAdd(&OUT[(size_t)segs[mloc] * D_FEAT + col], msg);
            } else {
                if (col < NCOL) OUT[(size_t)gr * NCOL + col] = val;
            }
        }
    }
}

// BN affine (inference running stats) + f32->f16, cols = 128.
template<bool HAS_BN>
__global__ __launch_bounds__(256) void bnact1(
    const float* __restrict__ X,
    const float* __restrict__ g,  const float* __restrict__ be,
    const float* __restrict__ mu, const float* __restrict__ var,
    _Float16* __restrict__ OUT, long n)
{
    long i = (long)blockIdx.x * blockDim.x + threadIdx.x;
    long stride = (long)gridDim.x * blockDim.x;
    for (; i < n; i += stride) {
        int k = (int)(i & (D_FEAT - 1));
        float v = X[i];
        if (HAS_BN) {
            float s = g[k] * rsqrtf(var[k] + 1e-3f);
            v = v * s + (be[k] - mu[k] * s);
        }
        OUT[i] = (_Float16)v;
    }
}

// Concat [Xa | Xb] (128+128) with 256-wide BN, f32->f16.
__global__ __launch_bounds__(256) void bnact2(
    const float* __restrict__ Xa, const float* __restrict__ Xb,
    const float* __restrict__ g,  const float* __restrict__ be,
    const float* __restrict__ mu, const float* __restrict__ var,
    _Float16* __restrict__ OUT, long rows)
{
    long n = rows * 256;
    long i = (long)blockIdx.x * blockDim.x + threadIdx.x;
    long stride = (long)gridDim.x * blockDim.x;
    for (; i < n; i += stride) {
        long r = i >> 8;
        int  k = (int)(i & 255);
        float v = (k < D_FEAT) ? Xa[r * D_FEAT + k] : Xb[r * D_FEAT + (k - D_FEAT)];
        float s = g[k] * rsqrtf(var[k] + 1e-3f);
        v = v * s + (be[k] - mu[k] * s);
        OUT[i] = (_Float16)v;
    }
}

// W [din][ncol] f32 -> Wt [doutp][din] f16, zero-padded cols >= ncol.
__global__ __launch_bounds__(256) void wconv_kernel(
    const float* __restrict__ W, _Float16* __restrict__ Wt,
    int din, int ncol, int doutp)
{
    long n = (long)doutp * din;
    long i = (long)blockIdx.x * blockDim.x + threadIdx.x;
    long stride = (long)gridDim.x * blockDim.x;
    for (; i < n; i += stride) {
        int nn = (int)(i / din), k = (int)(i % din);
        Wt[i] = (nn < ncol) ? (_Float16)W[(size_t)k * ncol + nn] : (_Float16)0.0f;
    }
}

// y = l2norm(y) + r, one wave32 per 128-float row, in place.
__global__ __launch_bounds__(256) void l2res_kernel(
    float* __restrict__ Y, const float* __restrict__ R, int nrows)
{
    int row = blockIdx.x * 8 + (threadIdx.x >> 5);
    if (row >= nrows) return;
    int lane = threadIdx.x & 31;
    float*       yp = Y + (size_t)row * D_FEAT + lane * 4;
    const float* rp = R + (size_t)row * D_FEAT + lane * 4;
    v4f y = *(const v4f*)yp;
    float ss = y[0]*y[0] + y[1]*y[1] + y[2]*y[2] + y[3]*y[3];
    #pragma unroll
    for (int m = 16; m >= 1; m >>= 1) ss += __shfl_xor(ss, m, 32);
    float inv = 1.0f / fmaxf(sqrtf(ss), 1e-12f);
    v4f r = *(const v4f*)rp;
    v4f o;
    #pragma unroll
    for (int i = 0; i < 4; ++i) o[i] = y[i] * inv + r[i];
    *(v4f*)yp = o;
}

__global__ void zero_kernel(float* p, long n) {
    long i = (long)blockIdx.x * blockDim.x + threadIdx.x;
    long stride = (long)gridDim.x * blockDim.x;
    for (; i < n; i += stride) p[i] = 0.0f;
}

__global__ __launch_bounds__(256) void sum_kernel(
    const float* __restrict__ w, float* __restrict__ out, int n)
{
    __shared__ float red[256];
    float s = 0.0f;
    for (int i = blockIdx.x * 256 + threadIdx.x; i < n; i += gridDim.x * 256)
        s += w[i];
    red[threadIdx.x] = s;
    __syncthreads();
    for (int off = 128; off > 0; off >>= 1) {
        if (threadIdx.x < off) red[threadIdx.x] += red[threadIdx.x + off];
        __syncthreads();
    }
    if (threadIdx.x == 0) atomicAdd(out, red[0]);
}

#define FP(i) ((const float*)d_in[(i)])

extern "C" void kernel_launch(void* const* d_in, const int* in_sizes, int n_in,
                              void* d_out, int out_size, void* d_ws, size_t ws_size,
                              hipStream_t stream) {
    const int D = 128;
    const int N = in_sizes[0] / D;
    const int E = in_sizes[2];
    const int B = in_sizes[3];
    const int C = in_sizes[41];

    const float* x_in     = (const float*)d_in[0];
    const int*   node_idx = (const int*)d_in[1];   // edges[0]
    const int*   nbr_idx  = node_idx + E;          // edges[1]
    const float* edge_w   = (const float*)d_in[2];
    const int*   inp_idx  = (const int*)d_in[3];

    // ---- workspace carve ----
    size_t ND = (size_t)N * D;
    char* p = (char*)d_ws;
    float* x0   = (float*)p;  p += ND * 4;
    float* xa   = (float*)p;  p += ND * 4;
    float* red  = (float*)p;  p += ND * 4;
    float* esum = (float*)p;  p += 16;
    _Float16* xh  = (_Float16*)p; p += ND * 2;            // [N,128] f16
    _Float16* xh2 = (_Float16*)p; p += (size_t)N * 256 * 2; // [N,256] f16
    _Float16* wt_pre = (_Float16*)p; p += 128 * 128 * 2;
    _Float16* wt_c1p = (_Float16*)p; p += 128 * 128 * 2;
    _Float16* wt_c1u = (_Float16*)p; p += 128 * 256 * 2;
    _Float16* wt_c2p = (_Float16*)p; p += 128 * 128 * 2;
    _Float16* wt_c2u = (_Float16*)p; p += 128 * 256 * 2;
    _Float16* wt_post = (_Float16*)p; p += 128 * 128 * 2;
    _Float16* wt_log  = (_Float16*)p; p += 48 * 128 * 2;

    dim3 blk(256);
    int gN = (N + 127) / 128;
    int gE = (E + 127) / 128;
    int gB = (B + 127) / 128;
    int gEl = 4096;   // elementwise grids
    int gW  = 64;

    // sum(edge_w)
    zero_kernel<<<1, 64, 0, stream>>>(esum, 1L);
    sum_kernel<<<256, blk, 0, stream>>>(edge_w, esum, E);

    // weights -> f16 transposed
    wconv_kernel<<<gW, blk, 0, stream>>>(FP(8),  wt_pre, 128, 128, 128);
    wconv_kernel<<<gW, blk, 0, stream>>>(FP(14), wt_c1p, 128, 128, 128);
    wconv_kernel<<<gW, blk, 0, stream>>>(FP(20), wt_c1u, 256, 128, 128);
    wconv_kernel<<<gW, blk, 0, stream>>>(FP(26), wt_c2p, 128, 128, 128);
    wconv_kernel<<<gW, blk, 0, stream>>>(FP(32), wt_c2u, 256, 128, 128);
    wconv_kernel<<<gW, blk, 0, stream>>>(FP(38), wt_post, 128, 128, 128);
    wconv_kernel<<<gW, blk, 0, stream>>>(FP(40), wt_log, 128, C, 48);

    // pre FFN: x0 = gelu(BN(nf) @ W + b)
    bnact1<true><<<gEl, blk, 0, stream>>>(x_in, FP(4), FP(5), FP(6), FP(7), xh, (long)ND);
    gemm_wmma<128, 128, false, 0><<<gN, blk, 0, stream>>>(
        xh, nullptr, wt_pre, FP(9), x0, nullptr, nullptr, nullptr, N, 128);

    // ---- conv 1 ----
    zero_kernel<<<2048, blk, 0, stream>>>(red, (long)ND);
    bnact1<true><<<gEl, blk, 0, stream>>>(x0, FP(10), FP(11), FP(12), FP(13), xh, (long)ND);
    gemm_wmma<128, 128, true, 1><<<gE, blk, 0, stream>>>(
        xh, nbr_idx, wt_c1p, FP(15), red, node_idx, edge_w, esum, E, 128);
    bnact2<<<2 * gEl, blk, 0, stream>>>(x0, red, FP(16), FP(17), FP(18), FP(19), xh2, (long)N);
    gemm_wmma<256, 128, false, 0><<<gN, blk, 0, stream>>>(
        xh2, nullptr, wt_c1u, FP(21), xa, nullptr, nullptr, nullptr, N, 128);
    l2res_kernel<<<(N + 7) / 8, blk, 0, stream>>>(xa, x0, N);   // xa = l2norm(xa)+x0

    // ---- conv 2 ----
    zero_kernel<<<2048, blk, 0, stream>>>(red, (long)ND);
    bnact1<true><<<gEl, blk, 0, stream>>>(xa, FP(22), FP(23), FP(24), FP(25), xh, (long)ND);
    gemm_wmma<128, 128, true, 1><<<gE, blk, 0, stream>>>(
        xh, nbr_idx, wt_c2p, FP(27), red, node_idx, edge_w, esum, E, 128);
    bnact2<<<2 * gEl, blk, 0, stream>>>(xa, red, FP(28), FP(29), FP(30), FP(31), xh2, (long)N);
    gemm_wmma<256, 128, false, 0><<<gN, blk, 0, stream>>>(
        xh2, nullptr, wt_c2u, FP(33), x0, nullptr, nullptr, nullptr, N, 128);
    l2res_kernel<<<(N + 7) / 8, blk, 0, stream>>>(x0, xa, N);   // x0 = l2norm(x0)+xa

    // post FFN: xa = gelu(BN(x0) @ W + b)
    bnact1<true><<<gEl, blk, 0, stream>>>(x0, FP(34), FP(35), FP(36), FP(37), xh, (long)ND);
    gemm_wmma<128, 128, false, 0><<<gN, blk, 0, stream>>>(
        xh, nullptr, wt_post, FP(39), xa, nullptr, nullptr, nullptr, N, 128);

    // logits head: out[B,C] = xa[input_node_idx] @ logits_w + logits_b
    bnact1<false><<<gEl, blk, 0, stream>>>(xa, nullptr, nullptr, nullptr, nullptr, xh, (long)ND);
    gemm_wmma<128, 48, true, 2><<<gB, blk, 0, stream>>>(
        xh, inp_idx, wt_log, FP(41), (float*)d_out, nullptr, nullptr, nullptr, B, C);
}
